// ApertisAttention_48361331753402
// MI455X (gfx1250) — compile-verified, matrix operand order checked by reading, not run on hardware
//
#include <hip/hip_runtime.h>

// ---------------------------------------------------------------------------
// Problem constants (ApertisAttention reference)
// ---------------------------------------------------------------------------
#define BB   2
#define SS   2048
#define HH   2048
#define NH   16
#define DD   128
#define MTOK (BB*SS)          // 4096 tokens
#define QSCALE 0.08838834764831845f  // 128^-0.5
#define LN_EPS 1e-12f

typedef __bf16 bf16x16 __attribute__((ext_vector_type(16)));
typedef __bf16 bf16x8  __attribute__((ext_vector_type(8)));
typedef float  f32x8   __attribute__((ext_vector_type(8)));
typedef unsigned int u32x4 __attribute__((ext_vector_type(4)));
typedef int          i32x4 __attribute__((ext_vector_type(4)));
typedef int          i32x8 __attribute__((ext_vector_type(8)));

// TDM (Tensor Data Mover) availability: device pass only; arity differs by
// toolchain (ROCm 7.2 clang-22 = 5 args; amdgpu-toolchain clang-23 = 6 args,
// discriminated by its TDM header shipping).
#if defined(__HIP_DEVICE_COMPILE__) && __has_builtin(__builtin_amdgcn_tensor_load_to_lds)
#define USE_TDM 1
#else
#define USE_TDM 0
#endif
#if USE_TDM && __has_include(<hip/amd_detail/amd_gfx1250_TDM.h>)
#define TDM_6ARG 1
#else
#define TDM_6ARG 0
#endif

__device__ __forceinline__ f32x8 wmma_bf16(bf16x16 a, bf16x16 b, f32x8 c) {
    return __builtin_amdgcn_wmma_f32_16x16x32_bf16(
        /*neg_a=*/false, a, /*neg_b=*/false, b,
        /*c_mod=*/(short)0, c, /*reuse_a=*/false, /*reuse_b=*/false);
}

// Load a 16x32 bf16 operand fragment (A layout; B-as-W^T uses the same
// addressing with rows indexed by N). Row-major source, leading dim `ld`.
// Per ISA 16-bit A layout: lane&15 = row, K chunks [kb..kb+7] and [kb+16..kb+23]
// with kb = (lane>>4)*8. Requires 16B alignment of each 8-elem chunk.
__device__ __forceinline__ bf16x16 load_frag(const __bf16* base, int ld) {
    int lane = threadIdx.x & 31;
    int r  = lane & 15;
    int kb = (lane >> 4) << 3;
    const bf16x8* p0 = (const bf16x8*)(base + (size_t)r * ld + kb);
    const bf16x8* p1 = (const bf16x8*)(base + (size_t)r * ld + kb + 16);
    bf16x8 lo = *p0;
    bf16x8 hi = *p1;
    bf16x16 f;
#pragma unroll
    for (int i = 0; i < 8; ++i) { f[i] = lo[i]; f[i + 8] = hi[i]; }
    return f;
}

// ---------------------------------------------------------------------------
// f32 -> bf16 convert (weights)
// ---------------------------------------------------------------------------
__global__ __launch_bounds__(256) void cvt_bf16_kernel(const float* __restrict__ in,
                                                       __bf16* __restrict__ out, int n) {
    int base = (blockIdx.x * 256 + threadIdx.x) * 8;
    if (base + 8 <= n) {
#pragma unroll
        for (int j = 0; j < 8; ++j) out[base + j] = (__bf16)in[base + j];
    }
}

// ---------------------------------------------------------------------------
// Fused LayerNorm chain: x = LN(h)*g+b ; the three per-projection LNs of x
// share mean/var -> emit xq/xk/xv bf16. One block per token, 256 thr, 8 f/thr.
// ---------------------------------------------------------------------------
__global__ __launch_bounds__(256) void ln_qkv_kernel(
    const float* __restrict__ hs,
    const float* __restrict__ ln_g,  const float* __restrict__ ln_b,
    const float* __restrict__ lnq_g, const float* __restrict__ lnq_b,
    const float* __restrict__ lnk_g, const float* __restrict__ lnk_b,
    const float* __restrict__ lnv_g, const float* __restrict__ lnv_b,
    __bf16* __restrict__ xq, __bf16* __restrict__ xk, __bf16* __restrict__ xv) {
    __shared__ float red0[8], red1[8];
    const int t    = blockIdx.x;
    const int tid  = threadIdx.x;
    const int lane = tid & 31, w = tid >> 5;
    const int base = tid * 8;
    const float* row = hs + (size_t)t * HH;

    float v[8];
#pragma unroll
    for (int i = 0; i < 8; ++i) v[i] = row[base + i];

    float s = 0.f, ss = 0.f;
#pragma unroll
    for (int i = 0; i < 8; ++i) { s += v[i]; ss += v[i] * v[i]; }
    for (int o = 16; o > 0; o >>= 1) { s += __shfl_xor(s, o, 32); ss += __shfl_xor(ss, o, 32); }
    if (lane == 0) { red0[w] = s; red1[w] = ss; }
    __syncthreads();
    float ts = 0.f, tss = 0.f;
#pragma unroll
    for (int j = 0; j < 8; ++j) { ts += red0[j]; tss += red1[j]; }
    float mu  = ts * (1.f / HH);
    float var = tss * (1.f / HH) - mu * mu;
    float rs  = rsqrtf(var + LN_EPS);

    float x[8];
#pragma unroll
    for (int i = 0; i < 8; ++i)
        x[i] = (v[i] - mu) * rs * ln_g[base + i] + ln_b[base + i];

    __syncthreads();
    s = 0.f; ss = 0.f;
#pragma unroll
    for (int i = 0; i < 8; ++i) { s += x[i]; ss += x[i] * x[i]; }
    for (int o = 16; o > 0; o >>= 1) { s += __shfl_xor(s, o, 32); ss += __shfl_xor(ss, o, 32); }
    if (lane == 0) { red0[w] = s; red1[w] = ss; }
    __syncthreads();
    ts = 0.f; tss = 0.f;
#pragma unroll
    for (int j = 0; j < 8; ++j) { ts += red0[j]; tss += red1[j]; }
    float mu2 = ts * (1.f / HH);
    float rs2 = rsqrtf(tss * (1.f / HH) - mu2 * mu2 + LN_EPS);

#pragma unroll
    for (int i = 0; i < 8; ++i) {
        int c = base + i;
        float xn = (x[i] - mu2) * rs2;
        size_t o = (size_t)t * HH + c;
        xq[o] = (__bf16)(xn * lnq_g[c] + lnq_b[c]);
        xk[o] = (__bf16)(xn * lnk_g[c] + lnk_b[c]);
        xv[o] = (__bf16)(xn * lnv_g[c] + lnv_b[c]);
    }
}

// ---------------------------------------------------------------------------
// bf16 GEMM: C[M,N] = A[M,K] @ W[N,K]^T (+bias) with f32 WMMA accumulate.
// Block 256 thr = 8 waves; tile 128x256x32; wave = 64x64 (4x4 WMMA tiles):
// 8 fragment loads per 16 WMMAs. LDS double-buffered: global loads for k+1
// are staged in VGPRs during the WMMA block for k; one barrier per k-step.
// mode 0: bf16 row-major out; mode 1: bf16 scatter to [B,NH,S,D] * scale;
// mode 2: f32 out = residual + C.
// ---------------------------------------------------------------------------
#define GBM 128
#define GBN 256
#define GBK 32
#define GLD 40   // padded LDS leading dim (32 + 8)

__global__ __launch_bounds__(256) void gemm_bf16_kernel(
    const __bf16* __restrict__ A, const __bf16* __restrict__ W,
    const float* __restrict__ bias,
    __bf16* __restrict__ outBf, float* __restrict__ outF32,
    const float* __restrict__ residual,
    int M, int N, int K, int mode, float scale) {
    __shared__ __bf16 As[2 * GBM * GLD];   // 2 x 128 x 40
    __shared__ __bf16 Bs[2 * GBN * GLD];   // 2 x 256 x 40

    const int tid  = threadIdx.x;
    const int lane = tid & 31;
    const int w    = tid >> 5;
    const int wr   = w >> 2;   // 0..1  (row half, 64 rows)
    const int wc   = w & 3;    // 0..3  (col quarter, 64 cols)
    const int m0   = blockIdx.y * GBM;
    const int n0   = blockIdx.x * GBN;

    f32x8 acc[4][4];
#pragma unroll
    for (int mt = 0; mt < 4; ++mt)
#pragma unroll
        for (int nt = 0; nt < 4; ++nt)
#pragma unroll
            for (int r = 0; r < 8; ++r) acc[mt][nt][r] = 0.f;

    // ---- preload k0 = 0 into buffer 0 ----
#pragma unroll
    for (int i = 0; i < 2; ++i) {
        int c = tid + 256 * i, row = c >> 2, col = (c & 3) << 3;
        bf16x8 a8 = *(const bf16x8*)(A + (size_t)(m0 + row) * K + col);
        *(bf16x8*)(As + row * GLD + col) = a8;
    }
#pragma unroll
    for (int i = 0; i < 4; ++i) {
        int c = tid + 256 * i, row = c >> 2, col = (c & 3) << 3;
        bf16x8 b8 = *(const bf16x8*)(W + (size_t)(n0 + row) * K + col);
        *(bf16x8*)(Bs + row * GLD + col) = b8;
    }
    __syncthreads();

    int cur = 0;
    for (int k0 = 0; k0 < K; k0 += GBK) {
        const bool hasNext = (k0 + GBK) < K;
        bf16x8 aReg[2], bReg[4];
        if (hasNext) {
            const int kn = k0 + GBK;
#pragma unroll
            for (int i = 0; i < 2; ++i) {
                int c = tid + 256 * i, row = c >> 2, col = (c & 3) << 3;
                aReg[i] = *(const bf16x8*)(A + (size_t)(m0 + row) * K + kn + col);
            }
#pragma unroll
            for (int i = 0; i < 4; ++i) {
                int c = tid + 256 * i, row = c >> 2, col = (c & 3) << 3;
                bReg[i] = *(const bf16x8*)(W + (size_t)(n0 + row) * K + kn + col);
            }
        }

        const __bf16* Ac = As + cur * (GBM * GLD);
        const __bf16* Bc = Bs + cur * (GBN * GLD);
        bf16x16 bfr[4];
#pragma unroll
        for (int nt = 0; nt < 4; ++nt)
            bfr[nt] = load_frag(Bc + (wc * 64 + nt * 16) * GLD, GLD);
#pragma unroll
        for (int mt = 0; mt < 4; ++mt) {
            bf16x16 af = load_frag(Ac + (wr * 64 + mt * 16) * GLD, GLD);
#pragma unroll
            for (int nt = 0; nt < 4; ++nt)
                acc[mt][nt] = wmma_bf16(af, bfr[nt], acc[mt][nt]);
        }

        if (hasNext) {
            const int nb = cur ^ 1;
            __bf16* An = As + nb * (GBM * GLD);
            __bf16* Bn = Bs + nb * (GBN * GLD);
#pragma unroll
            for (int i = 0; i < 2; ++i) {
                int c = tid + 256 * i;
                *(bf16x8*)(An + (c >> 2) * GLD + ((c & 3) << 3)) = aReg[i];
            }
#pragma unroll
            for (int i = 0; i < 4; ++i) {
                int c = tid + 256 * i;
                *(bf16x8*)(Bn + (c >> 2) * GLD + ((c & 3) << 3)) = bReg[i];
            }
        }
        __syncthreads();
        cur ^= 1;
    }

    // epilogue: C layout is VGPR r -> row (r + 8*(lane>=16)), col = lane&15
#pragma unroll
    for (int mt = 0; mt < 4; ++mt) {
#pragma unroll
        for (int nt = 0; nt < 4; ++nt) {
            int col = n0 + wc * 64 + nt * 16 + (lane & 15);
            float bv = bias[col];
#pragma unroll
            for (int r = 0; r < 8; ++r) {
                int row = m0 + wr * 64 + mt * 16 + r + ((lane >> 4) << 3);
                float val = (acc[mt][nt][r] + bv) * scale;
                if (mode == 0) {
                    outBf[(size_t)row * N + col] = (__bf16)val;
                } else if (mode == 1) {
                    int b = row >> 11, sTok = row & (SS - 1);
                    int h = col >> 7, d = col & (DD - 1);
                    outBf[((((size_t)b * NH + h) * SS + sTok) * DD) + d] = (__bf16)val;
                } else {
                    size_t idx = (size_t)row * N + col;
                    outF32[idx] = residual[idx] + val;
                }
            }
        }
    }
}

// ---------------------------------------------------------------------------
// Fused flash attention: per (q-tile of 128 rows, b*h). 8 waves, 16 q-rows
// each; K/V tiles of 64 stream through LDS; online softmax in VGPRs; WMMA for
// both Q@K^T and P@V. K tile is fetched by the Tensor Data Mover (TDM) with
// hardware LDS row padding (64 data DWORDs + 4 pad DWORDs = 136-elem ld),
// tracked by TENSORcnt; falls back to manual loads if builtin unavailable.
// ctx written token-major [B*S, H] bf16 for the Wo GEMM.
// ---------------------------------------------------------------------------
#define AKV 64
#define KLD 136   // 128 + 8
#define VLD 72    // 64 + 8
#define PLD 72

__global__ __launch_bounds__(256) void attn_kernel(
    const __bf16* __restrict__ Q, const __bf16* __restrict__ Km,
    const __bf16* __restrict__ Vm, __bf16* __restrict__ ctx) {
    __shared__ __bf16 Ks[AKV * KLD];     // [kv=64][d=128] (+pad)
    __shared__ __bf16 VTs[DD * VLD];     // [d=128][kv=64] (+pad)
    __shared__ __bf16 Ps[8 * 16 * PLD];  // per-wave 16 x 64 P scratch

    const int tid  = threadIdx.x;
    const int lane = tid & 31;
    const int w    = tid >> 5;
    const int bh   = blockIdx.y;
    const int b    = bh >> 4;
    const int h    = bh & (NH - 1);
    const size_t head_off = (size_t)bh * SS * DD;

    const __bf16* Qb = Q  + head_off;
    const __bf16* Kb = Km + head_off;
    const __bf16* Vb = Vm + head_off;

    const int qrow0 = blockIdx.x * 128 + w * 16;

    // Q fragments for this wave: 16 x 128 = 4 chunks of 16x32, held in VGPRs
    bf16x16 qf[4];
#pragma unroll
    for (int kc = 0; kc < 4; ++kc)
        qf[kc] = load_frag(Qb + (size_t)qrow0 * DD + kc * 32, DD);

    f32x8 o[8];
#pragma unroll
    for (int nt = 0; nt < 8; ++nt)
#pragma unroll
        for (int r = 0; r < 8; ++r) o[nt][r] = 0.f;
    float mst[8], lst[8];
#pragma unroll
    for (int r = 0; r < 8; ++r) { mst[r] = -3.0e38f; lst[r] = 0.f; }

    __bf16* Pw = Ps + w * 16 * PLD;

    for (int kt = 0; kt < SS; kt += AKV) {
#if USE_TDM
        // --- TDM: DMA the 64x128 bf16 K tile into LDS with row padding ---
        if (w == 0) {
            unsigned long long ga = (unsigned long long)(Kb + (size_t)kt * DD);
            unsigned int lds = (unsigned int)(unsigned long long)(&Ks[0]);
            u32x4 g0 = { 1u,                                   // count=1
                         lds,                                  // lds_addr
                         (unsigned int)ga,                     // global_addr[31:0]
                         (unsigned int)((ga >> 32) & 0x1ffffffu) | (2u << 30) }; // [56:32] | type=2
            i32x8 g1 = { (1 << 16) | (1 << 20) | (5 << 22) | (3 << 25), // 2B elems, pad_en, 64 dw interval, 4 dw pad
                         (DD & 0xffff) << 16,                  // tensor_dim0 lo
                         (SS & 0xffff) << 16,                  // tensor_dim0 hi | tensor_dim1 lo
                         (DD & 0xffff) << 16,                  // tensor_dim1 hi | tile_dim0 = 128
                         AKV,                                  // tile_dim1 = 64
                         DD,                                   // tensor_dim0_stride = 128
                         0, 0 };
            i32x4 gz = { 0, 0, 0, 0 };
#if TDM_6ARG
            i32x8 gz8 = { 0, 0, 0, 0, 0, 0, 0, 0 };
            __builtin_amdgcn_tensor_load_to_lds(g0, g1, gz, gz, gz8, 0);
#else
            __builtin_amdgcn_tensor_load_to_lds(g0, g1, gz, gz, 0);
#endif
        }
#endif
        // cooperative V tile transpose into LDS (and K tile if no TDM)
#pragma unroll
        for (int i = 0; i < 4; ++i) {
            int c   = tid + 256 * i;       // 1024 chunks of 8 elems
            int row = c >> 4;              // 0..63 (kv)
            int col = (c & 15) << 3;       // 0..120 (d)
#if !USE_TDM
            bf16x8 k8 = *(const bf16x8*)(Kb + (size_t)(kt + row) * DD + col);
            *(bf16x8*)(Ks + row * KLD + col) = k8;
#endif
            bf16x8 v8 = *(const bf16x8*)(Vb + (size_t)(kt + row) * DD + col);
#pragma unroll
            for (int j = 0; j < 8; ++j) VTs[(col + j) * VLD + row] = v8[j];
        }
#if USE_TDM
        __builtin_amdgcn_s_wait_tensorcnt(0);  // wave 0 waits; others trivially pass
#endif
        __syncthreads();

        // scores S = Q(16x128) @ K^T(128x64): 4 n-tiles x 4 k-chunks
        f32x8 sc[4];
#pragma unroll
        for (int nt = 0; nt < 4; ++nt) {
#pragma unroll
            for (int r = 0; r < 8; ++r) sc[nt][r] = 0.f;
#pragma unroll
            for (int kc = 0; kc < 4; ++kc) {
                bf16x16 kf = load_frag(Ks + (nt * 16) * KLD + kc * 32, KLD);
                sc[nt] = wmma_bf16(qf[kc], kf, sc[nt]);
            }
        }

        // online softmax per row (row r + 8*(lane>=16); cols across lanes&15)
#pragma unroll
        for (int r = 0; r < 8; ++r) {
            float tmax = sc[0][r];
#pragma unroll
            for (int nt = 1; nt < 4; ++nt) tmax = fmaxf(tmax, sc[nt][r]);
            for (int oI = 8; oI > 0; oI >>= 1) tmax = fmaxf(tmax, __shfl_xor(tmax, oI, 16));
            float mnew = fmaxf(mst[r], tmax);
            float corr = __expf(mst[r] - mnew);
            float rsum = 0.f;
#pragma unroll
            for (int nt = 0; nt < 4; ++nt) {
                float p = __expf(sc[nt][r] - mnew);
                rsum += p;
                int rowP = r + ((lane >> 4) << 3);
                int colP = nt * 16 + (lane & 15);
                Pw[rowP * PLD + colP] = (__bf16)p;
            }
            for (int oI = 8; oI > 0; oI >>= 1) rsum += __shfl_xor(rsum, oI, 16);
            lst[r] = lst[r] * corr + rsum;
            mst[r] = mnew;
#pragma unroll
            for (int nt = 0; nt < 8; ++nt) o[nt][r] *= corr;
        }

        // O += P(16x64) @ V(64x128): 2 k-chunks x 8 n-tiles (VT gives B layout)
#pragma unroll
        for (int kc = 0; kc < 2; ++kc) {
            bf16x16 pf = load_frag(Pw + kc * 32, PLD);
#pragma unroll
            for (int nt = 0; nt < 8; ++nt) {
                bf16x16 vf = load_frag(VTs + (nt * 16) * VLD + kc * 32, VLD);
                o[nt] = wmma_bf16(pf, vf, o[nt]);
            }
        }
        __syncthreads();
    }

    // epilogue: ctx token-major [B*S][H], bf16
#pragma unroll
    for (int nt = 0; nt < 8; ++nt) {
        int d = nt * 16 + (lane & 15);
#pragma unroll
        for (int r = 0; r < 8; ++r) {
            int sTok = qrow0 + r + ((lane >> 4) << 3);
            float val = o[nt][r] / lst[r];
            ctx[((size_t)b * SS + sTok) * HH + h * DD + d] = (__bf16)val;
        }
    }
}

// ---------------------------------------------------------------------------
// Host-side launcher
// ---------------------------------------------------------------------------
extern "C" void kernel_launch(void* const* d_in, const int* in_sizes, int n_in,
                              void* d_out, int out_size, void* d_ws, size_t ws_size,
                              hipStream_t stream) {
    const float* hs    = (const float*)d_in[0];
    const float* ln_g  = (const float*)d_in[1];
    const float* ln_b  = (const float*)d_in[2];
    const float* lnq_g = (const float*)d_in[3];
    const float* lnq_b = (const float*)d_in[4];
    const float* lnk_g = (const float*)d_in[5];
    const float* lnk_b = (const float*)d_in[6];
    const float* lnv_g = (const float*)d_in[7];
    const float* lnv_b = (const float*)d_in[8];
    const float* Wq = (const float*)d_in[9];  const float* bq = (const float*)d_in[10];
    const float* Wk = (const float*)d_in[11]; const float* bk = (const float*)d_in[12];
    const float* Wv = (const float*)d_in[13]; const float* bv = (const float*)d_in[14];
    const float* Wo = (const float*)d_in[15]; const float* bo = (const float*)d_in[16];
    const float* Wout = (const float*)d_in[17]; const float* bout = (const float*)d_in[18];
    float* out = (float*)d_out;

    char* ws = (char*)d_ws;
    size_t off = 0;
    auto alloc = [&](size_t bytes) -> void* {
        void* p = ws + off;
        off += (bytes + 255) & ~(size_t)255;
        return p;
    };
    const size_t WBYTES = (size_t)HH * HH * sizeof(__bf16);         // 8 MB
    const size_t XBYTES = (size_t)MTOK * HH * sizeof(__bf16);       // 16 MB
    __bf16* wq  = (__bf16*)alloc(WBYTES);
    __bf16* wk  = (__bf16*)alloc(WBYTES);
    __bf16* wv  = (__bf16*)alloc(WBYTES);
    __bf16* wo  = (__bf16*)alloc(WBYTES);
    __bf16* wou = (__bf16*)alloc(WBYTES);
    __bf16* xq  = (__bf16*)alloc(XBYTES);
    __bf16* xk  = (__bf16*)alloc(XBYTES);
    __bf16* xv  = (__bf16*)alloc(XBYTES);
    __bf16* qb  = (__bf16*)alloc(XBYTES);   // [B,NH,S,D]
    __bf16* kb2 = (__bf16*)alloc(XBYTES);
    __bf16* vb2 = (__bf16*)alloc(XBYTES);
    __bf16* ctx = (__bf16*)alloc(XBYTES);   // token-major
    __bf16* tmp = (__bf16*)alloc(XBYTES);

    // 1) weights -> bf16
    const int wn = HH * HH;
    cvt_bf16_kernel<<<wn / (256 * 8), 256, 0, stream>>>(Wq, wq, wn);
    cvt_bf16_kernel<<<wn / (256 * 8), 256, 0, stream>>>(Wk, wk, wn);
    cvt_bf16_kernel<<<wn / (256 * 8), 256, 0, stream>>>(Wv, wv, wn);
    cvt_bf16_kernel<<<wn / (256 * 8), 256, 0, stream>>>(Wo, wo, wn);
    cvt_bf16_kernel<<<wn / (256 * 8), 256, 0, stream>>>(Wout, wou, wn);

    // 2) fused LN chain -> xq/xk/xv (bf16)
    ln_qkv_kernel<<<MTOK, 256, 0, stream>>>(hs, ln_g, ln_b, lnq_g, lnq_b,
                                            lnk_g, lnk_b, lnv_g, lnv_b,
                                            xq, xk, xv);

    // 3) q/k/v projections: scatter to [B,NH,S,D]; q pre-scaled by D^-0.5
    dim3 gg(HH / GBN, MTOK / GBM);  // (8, 32)
    gemm_bf16_kernel<<<gg, 256, 0, stream>>>(xq, wq, bq, qb,  nullptr, nullptr,
                                             MTOK, HH, HH, 1, QSCALE);
    gemm_bf16_kernel<<<gg, 256, 0, stream>>>(xk, wk, bk, kb2, nullptr, nullptr,
                                             MTOK, HH, HH, 1, 1.0f);
    gemm_bf16_kernel<<<gg, 256, 0, stream>>>(xv, wv, bv, vb2, nullptr, nullptr,
                                             MTOK, HH, HH, 1, 1.0f);

    // 4) fused flash attention -> ctx (token-major bf16)
    attn_kernel<<<dim3(SS / 128, BB * NH), 256, 0, stream>>>(qb, kb2, vb2, ctx);

    // 5) o_proj: tmp = ctx @ Wo^T + bo
    gemm_bf16_kernel<<<gg, 256, 0, stream>>>(ctx, wo, bo, tmp, nullptr, nullptr,
                                             MTOK, HH, HH, 0, 1.0f);

    // 6) output projection + residual: out = hs + tmp @ Wout^T + bout (f32)
    gemm_bf16_kernel<<<gg, 256, 0, stream>>>(tmp, wou, bout, nullptr, out, hs,
                                             MTOK, HH, HH, 2, 1.0f);
}
